// Sepconv_55568286875971
// MI455X (gfx1250) — matrix operand order repaired and checked by reading, not compiled, not run on hardware
//
#include <hip/hip_runtime.h>
#include <hip/hip_bf16.h>

typedef _Float16 h16;
typedef __attribute__((ext_vector_type(16))) _Float16 v16h;
typedef __attribute__((ext_vector_type(8)))  float    v8f;
typedef __attribute__((ext_vector_type(4)))  unsigned int u32x4;

union ABfrag { u32x4 q[2]; v16h h; };

// ---------------------------------------------------------------------------
// Weight repack: fp32 OIHW -> f16 WMMA B-fragment-linear layout.
// Fragment index: (((t*KB + kb)*NB + nb)*32 + lane)*16 + j
// with K = kb*32 + j + 16*(lane>>4),  N = nb*16 + (lane&15).
// Out-of-range (Cin/Cout padding) -> 0 so padded channels stay exactly zero.
// ---------------------------------------------------------------------------
__global__ __launch_bounds__(256) void repack_w(const float* __restrict__ w,
                                                h16* __restrict__ wp,
                                                int Cin, int Cout, int KB, int NB) {
  long idx = (long)blockIdx.x * 256 + threadIdx.x;
  long total = (long)9 * KB * NB * 512;
  if (idx >= total) return;
  int j    = idx & 15;
  int lane = (idx >> 4) & 31;
  long f   = idx >> 9;
  int nb   = (int)(f % NB);
  long f2  = f / NB;
  int kb   = (int)(f2 % KB);
  int t    = (int)(f2 / KB);
  int k = kb * 32 + j + 16 * (lane >> 4);
  int n = nb * 16 + (lane & 15);
  float v = 0.f;
  if (k < Cin && n < Cout) v = w[((size_t)n * Cin + k) * 9 + t];
  wp[idx] = (h16)v;
}

// ---------------------------------------------------------------------------
// Implicit-GEMM 3x3 conv (pad=1), NHWC f16 activations (channels padded to
// 32), f32 WMMA accumulation, fused bias + {none|relu|tanh}.
// One wave = 16 pixels x (16*NACC) couts: A fragment loaded once per
// (tap, k-block) and reused for NACC WMMAs with contiguous B fragments.
// 8 waves / 256-thread block. 9-tap loop fully unrolled (const dy/dx).
// Boundary handling: clamped address + cndmask-zero (no exec divergence).
// ---------------------------------------------------------------------------
template <int NACC>
__global__ __launch_bounds__(256) void conv3x3_wmma(
    const h16* __restrict__ in, const h16* __restrict__ wp,
    const float* __restrict__ bias, h16* __restrict__ out,
    int H, int W, int Cinp, int Coutp, int Cout, int MT, int NT4, int NB,
    int postop) {
  int wave = threadIdx.x >> 5;
  int lane = threadIdx.x & 31;
  int tile = blockIdx.x * 8 + wave;
  if (tile >= MT * NT4) return;
  int mt = tile % MT;
  int nbBase = (tile / MT) * NACC;
  int pixbase = mt * 16;
  int p = pixbase + (lane & 15);
  int y = p / W, x = p % W;
  int KB = Cinp >> 5;
  int kofs = 16 * (lane >> 4);  // A-fragment per-lane channel sub-offset
  u32x4 zero = {0u, 0u, 0u, 0u};
  v8f acc[NACC] = {};
#pragma unroll
  for (int t = 0; t < 9; ++t) {
    const int dy = t / 3 - 1, dx = t % 3 - 1;
    int yy = y + dy, xx = x + dx;
    bool inb = ((unsigned)yy < (unsigned)H) && ((unsigned)xx < (unsigned)W);
    int yyc = min(max(yy, 0), H - 1);
    int xxc = min(max(xx, 0), W - 1);
    const h16* arow = in + (size_t)(yyc * W + xxc) * Cinp + kofs;
    const h16* wrow = wp + (((size_t)t * KB) * NB + nbBase) * 512 + (size_t)lane * 16;
    for (int kb = 0; kb < KB; ++kb) {
      ABfrag a;
      u32x4 q0 = *reinterpret_cast<const u32x4*>(arow);
      u32x4 q1 = *reinterpret_cast<const u32x4*>(arow + 8);
      a.q[0] = inb ? q0 : zero;
      a.q[1] = inb ? q1 : zero;
      __builtin_prefetch(wrow + (size_t)NB * 512, 0, 1);  // next k-block's weights
#pragma unroll
      for (int u = 0; u < NACC; ++u) {
        ABfrag b;
        const h16* wfrag = wrow + (size_t)u * 512;
        b.q[0] = *reinterpret_cast<const u32x4*>(wfrag);
        b.q[1] = *reinterpret_cast<const u32x4*>(wfrag + 8);
        acc[u] = __builtin_amdgcn_wmma_f32_16x16x32_f16(
            false, a.h, false, b.h, (short)0, acc[u], false, false);
      }
      arow += 32;
      wrow += (size_t)NB * 512;
    }
  }
  int n = lane & 15;
  int rofs = 8 * (lane >> 4);
#pragma unroll
  for (int u = 0; u < NACC; ++u) {
    int col = (nbBase + u) * 16 + n;
    float bv = (col < Cout) ? bias[col] : 0.f;
    for (int r = 0; r < 8; ++r) {
      float v = acc[u][r] + bv;
      if (postop == 1) v = fmaxf(v, 0.f);
      else if (postop == 2) v = tanhf(v);
      out[(size_t)(pixbase + rofs + r) * Coutp + col] = (h16)v;
    }
  }
}

// ---------------------------------------------------------------------------
// Elementwise helpers (NHWC f16, padded channel count Cp).
// ---------------------------------------------------------------------------
__global__ __launch_bounds__(256) void pack_inputs(const float* __restrict__ a,
                                                   const float* __restrict__ b,
                                                   h16* __restrict__ out) {
  int p = blockIdx.x * 256 + threadIdx.x;      // 65536 pixels
  h16* o = out + (size_t)p * 32;
  for (int c = 0; c < 3; ++c) { o[c] = (h16)a[c * 65536 + p]; o[3 + c] = (h16)b[c * 65536 + p]; }
  for (int c = 6; c < 32; ++c) o[c] = (h16)0.f;
}

__global__ __launch_bounds__(256) void avgpool2k(const h16* __restrict__ in,
                                                 h16* __restrict__ out,
                                                 int Ho, int Wo, int Cp, long total) {
  long idx = (long)blockIdx.x * 256 + threadIdx.x;
  if (idx >= total) return;
  int c = (int)(idx % Cp);
  long pp = idx / Cp;
  int x = (int)(pp % Wo), y = (int)(pp / Wo);
  int Wi = Wo * 2;
  const h16* i0 = in + ((size_t)(2 * y) * Wi + 2 * x) * Cp + c;
  const h16* i1 = i0 + (size_t)Wi * Cp;
  float s = (float)i0[0] + (float)i0[Cp] + (float)i1[0] + (float)i1[Cp];
  out[idx] = (h16)(0.25f * s);
}

__global__ __launch_bounds__(256) void up2_bilinear(const h16* __restrict__ in,
                                                    h16* __restrict__ out,
                                                    int Hi, int Wi, int Cp, long total) {
  long idx = (long)blockIdx.x * 256 + threadIdx.x;
  if (idx >= total) return;
  int c = (int)(idx % Cp);
  long pp = idx / Cp;
  int Wo = Wi * 2;
  int X = (int)(pp % Wo), Y = (int)(pp / Wo);
  float sy = 0.5f * (Y + 0.5f) - 0.5f;
  float sx = 0.5f * (X + 0.5f) - 0.5f;
  int y0 = (int)floorf(sy), x0 = (int)floorf(sx);
  float fy = sy - y0, fx = sx - x0;
  int y0c = min(max(y0, 0), Hi - 1), y1c = min(max(y0 + 1, 0), Hi - 1);
  int x0c = min(max(x0, 0), Wi - 1), x1c = min(max(x0 + 1, 0), Wi - 1);
  const h16* base = in + c;
  float v00 = (float)base[((size_t)y0c * Wi + x0c) * Cp];
  float v01 = (float)base[((size_t)y0c * Wi + x1c) * Cp];
  float v10 = (float)base[((size_t)y1c * Wi + x0c) * Cp];
  float v11 = (float)base[((size_t)y1c * Wi + x1c) * Cp];
  float v = (1.f - fy) * ((1.f - fx) * v00 + fx * v01) + fy * ((1.f - fx) * v10 + fx * v11);
  out[idx] = (h16)v;
}

__global__ __launch_bounds__(256) void addk(const h16* __restrict__ a,
                                            const h16* __restrict__ b,
                                            h16* __restrict__ out, long n) {
  long idx = (long)blockIdx.x * 256 + threadIdx.x;
  if (idx >= n) return;
  out[idx] = (h16)((float)a[idx] + (float)b[idx]);
}

// ---------------------------------------------------------------------------
// Separable-conv application: out[c,y,x] (+)= sum_{v,h} kv[v]*kh[h]*P[y+v,x+h]
// with edge-padded frame (PAD=25). 16x16 pixel tile per block; 66x66 fp32
// channel tile staged in LDS via GLOBAL_LOAD_ASYNC_TO_LDS_B32 (ASYNCcnt);
// 51 kh taps kept in registers (contiguous f16 loads).
// LDS byte offset = low 32 bits of the generic shared-pointer (aperture rule).
// ---------------------------------------------------------------------------
__device__ __forceinline__ void async_g2l_b32(unsigned lds_off, const void* gaddr) {
  unsigned long long ga = (unsigned long long)gaddr;
  asm volatile("global_load_async_to_lds_b32 %0, %1, off"
               :: "v"(lds_off), "v"(ga) : "memory");
}
__device__ __forceinline__ void wait_asynccnt0() {
#if __has_builtin(__builtin_amdgcn_s_wait_asynccnt)
  __builtin_amdgcn_s_wait_asynccnt(0);
#else
  asm volatile("s_wait_asynccnt 0x0" ::: "memory");
#endif
}

__global__ __launch_bounds__(256) void sepconv_apply(
    const float* __restrict__ frame, const h16* __restrict__ kv,
    const h16* __restrict__ kh, float* __restrict__ out, int accumulate) {
  __shared__ float tile[66 * 66];
  int tx = threadIdx.x & 15, ty = threadIdx.x >> 4;
  int x0 = blockIdx.x * 16, y0 = blockIdx.y * 16;
  int x = x0 + tx, y = y0 + ty;
  int pix = y * 256 + x;
  float khr[51];
  const h16* khp = kh + (size_t)pix * 64;
  for (int h = 0; h < 51; ++h) khr[h] = (float)khp[h];
  const h16* kvp = kv + (size_t)pix * 64;
  for (int c = 0; c < 3; ++c) {
    __syncthreads();
    const float* fc = frame + (size_t)c * 65536;
    for (int i = threadIdx.x; i < 66 * 66; i += 256) {
      int r = i / 66, cc = i % 66;
      int gy = min(max(y0 - 25 + r, 0), 255);
      int gx = min(max(x0 - 25 + cc, 0), 255);
      async_g2l_b32((unsigned)(size_t)(&tile[i]), fc + gy * 256 + gx);
    }
    wait_asynccnt0();
    __syncthreads();
    float acc = 0.f;
    for (int v = 0; v < 51; ++v) {
      float kvv = (float)kvp[v];
      const float* row = &tile[(ty + v) * 66 + tx];
      float s = 0.f;
#pragma unroll 17
      for (int h = 0; h < 51; ++h) s = fmaf(khr[h], row[h], s);
      acc = fmaf(kvv, s, acc);
    }
    float* op = out + (size_t)c * 65536 + pix;
    if (accumulate) *op += acc; else *op = acc;
  }
}

// ---------------------------------------------------------------------------
// Host orchestration
// ---------------------------------------------------------------------------
extern "C" void kernel_launch(void* const* d_in, const int* in_sizes, int n_in,
                              void* d_out, int out_size, void* d_ws, size_t ws_size,
                              hipStream_t stream) {
  (void)in_sizes; (void)n_in; (void)out_size; (void)ws_size;
  const float* in1 = (const float*)d_in[0];
  const float* in2 = (const float*)d_in[1];
  float* outp = (float*)d_out;

  auto F = [&](int i) { return (const float*)d_in[i]; };
  auto cdiv = [](long a, long b) { return (a + b - 1) / b; };
  auto c32 = [](int c) { return (c + 31) & ~31; };

  // bump allocator in d_ws
  char* base = (char*)d_ws;
  size_t off = 0;
  auto allocH = [&](size_t nelem) -> h16* {
    h16* p = (h16*)(base + off);
    off = (off + nelem * sizeof(h16) + 255) & ~(size_t)255;
    return p;
  };
  h16* tA    = allocH((size_t)65536 * 64);
  h16* tB    = allocH((size_t)65536 * 64);
  h16* tC    = allocH((size_t)65536 * 64);
  h16* x0    = allocH((size_t)65536 * 32);
  h16* c2b   = allocH((size_t)16384 * 64);
  h16* c3b   = allocH((size_t)4096 * 128);
  h16* c4b   = allocH((size_t)1024 * 256);
  h16* c5b   = allocH((size_t)256 * 512);
  h16* comb9 = allocH((size_t)16384 * 64);
  h16* kbuf[4];
  for (int i = 0; i < 4; ++i) kbuf[i] = allocH((size_t)65536 * 64);
  h16* wpack = allocH((size_t)9 * 16 * 32 * 512);

  auto conv = [&](const h16* in, int H, int W, int Cin, int Cout,
                  int widx, h16* out, int postop) {
    int Cinp = c32(Cin), Coutp = c32(Cout);
    int KB = Cinp / 32, NB = Coutp / 16;
    long wtotal = (long)9 * KB * NB * 512;
    repack_w<<<(unsigned)cdiv(wtotal, 256), 256, 0, stream>>>(F(widx), wpack, Cin, Cout, KB, NB);
    int MT = H * W / 16;
    int NACC = (NB % 4 == 0) ? 4 : 2;
    int NT4 = NB / NACC;
    long tiles = (long)MT * NT4;
    unsigned blocks = (unsigned)cdiv(tiles, 8);
    if (NACC == 4) {
      conv3x3_wmma<4><<<blocks, 256, 0, stream>>>(
          in, wpack, F(widx + 1), out, H, W, Cinp, Coutp, Cout, MT, NT4, NB, postop);
    } else {
      conv3x3_wmma<2><<<blocks, 256, 0, stream>>>(
          in, wpack, F(widx + 1), out, H, W, Cinp, Coutp, Cout, MT, NT4, NB, postop);
    }
  };
  auto pool = [&](const h16* in, h16* out, int Ho, int Wo, int Cp) {
    long n = (long)Ho * Wo * Cp;
    avgpool2k<<<(unsigned)cdiv(n, 256), 256, 0, stream>>>(in, out, Ho, Wo, Cp, n);
  };
  auto up2 = [&](const h16* in, h16* out, int Hi, int Wi, int Cp) {
    long n = (long)4 * Hi * Wi * Cp;
    up2_bilinear<<<(unsigned)cdiv(n, 256), 256, 0, stream>>>(in, out, Hi, Wi, Cp, n);
  };
  auto add = [&](const h16* a, const h16* b, h16* out, long n) {
    addk<<<(unsigned)cdiv(n, 256), 256, 0, stream>>>(a, b, out, n);
  };

  const int RELU = 1, TANH = 2;

  pack_inputs<<<256, 256, 0, stream>>>(in1, in2, x0);

  // encoder
  conv(x0, 256, 256,   6,  32,  2, tA, RELU);
  conv(tA, 256, 256,  32,  32,  4, tB, RELU);
  conv(tB, 256, 256,  32,  32,  6, tA, RELU);          // c1
  pool(tA, tB, 128, 128, 32);
  conv(tB, 128, 128,  32,  64,  8, tA, RELU);
  conv(tA, 128, 128,  64,  64, 10, tB, RELU);
  conv(tB, 128, 128,  64,  64, 12, c2b, RELU);         // c2
  pool(c2b, tA, 64, 64, 64);
  conv(tA,  64,  64,  64, 128, 14, tB, RELU);
  conv(tB,  64,  64, 128, 128, 16, tA, RELU);
  conv(tA,  64,  64, 128, 128, 18, c3b, RELU);         // c3
  pool(c3b, tA, 32, 32, 128);
  conv(tA,  32,  32, 128, 256, 20, tB, RELU);
  conv(tB,  32,  32, 256, 256, 22, tA, RELU);
  conv(tA,  32,  32, 256, 256, 24, c4b, RELU);         // c4
  pool(c4b, tA, 16, 16, 256);
  conv(tA,  16,  16, 256, 512, 26, tB, RELU);
  conv(tB,  16,  16, 512, 512, 28, tA, RELU);
  conv(tA,  16,  16, 512, 512, 30, c5b, RELU);         // c5
  pool(c5b, tA, 8, 8, 512);
  conv(tA,   8,   8, 512, 512, 32, tB, RELU);
  conv(tB,   8,   8, 512, 512, 34, tA, RELU);
  conv(tA,   8,   8, 512, 512, 36, tB, RELU);          // c6

  // decoder
  up2(tB, tA, 8, 8, 512);
  conv(tA, 16, 16, 512, 512, 80, tB, RELU);            // up6
  add(tB, c5b, tA, (long)256 * 512);
  conv(tA, 16, 16, 512, 256, 38, tB, RELU);
  conv(tB, 16, 16, 256, 256, 40, tA, RELU);
  conv(tA, 16, 16, 256, 256, 42, tB, RELU);            // c7
  up2(tB, tA, 16, 16, 256);
  conv(tA, 32, 32, 256, 256, 82, tB, RELU);            // up7
  add(tB, c4b, tA, (long)1024 * 256);
  conv(tA, 32, 32, 256, 128, 44, tB, RELU);
  conv(tB, 32, 32, 128, 128, 46, tA, RELU);
  conv(tA, 32, 32, 128, 128, 48, tB, RELU);            // c8
  up2(tB, tA, 32, 32, 128);
  conv(tA, 64, 64, 128, 128, 84, tB, RELU);            // up8
  add(tB, c3b, tA, (long)4096 * 128);
  conv(tA, 64, 64, 128,  64, 50, tB, RELU);
  conv(tB, 64, 64,  64,  64, 52, tA, RELU);
  conv(tA, 64, 64,  64,  64, 54, tB, RELU);            // c9
  up2(tB, tA, 64, 64, 64);
  conv(tA, 128, 128, 64, 64, 86, tB, RELU);            // up9
  add(tB, c2b, comb9, (long)16384 * 64);               // comb9

  // four kernel-prediction branches: (front, back, out)
  const int fidx[4] = {88, 56, 92, 60};  // vert1, horz1, vert2, horz2
  const int bidx[4] = {72, 64, 76, 68};  // kv1,   kh1,   kv2,   kh2
  for (int d = 0; d < 4; ++d) {
    conv(comb9, 128, 128, 64, 64, fidx[d],     tA, RELU);
    conv(tA,    128, 128, 64, 64, fidx[d] + 2, tB, RELU);
    conv(tB,    128, 128, 64, 51, bidx[d],     tA, RELU);   // 64 -> 51 (pad 64)
    up2(tA, tC, 128, 128, 64);
    conv(tC,    256, 256, 51, 51, bidx[d] + 2, kbuf[d], TANH);
  }

  // separable conv application + sum of both frames
  dim3 grid(16, 16);
  sepconv_apply<<<grid, 256, 0, stream>>>(in1, kbuf[0], kbuf[1], outp, 0);
  sepconv_apply<<<grid, 256, 0, stream>>>(in2, kbuf[2], kbuf[3], outp, 1);
}
